// CrossAttention_20684562498086
// MI455X (gfx1250) — matrix operand order repaired and checked by reading, not compiled, not run on hardware
//
#include <hip/hip_runtime.h>
#include <hip/hip_bf16.h>
#include <math.h>

// ---------------- types ----------------
typedef _Float16 h8  __attribute__((ext_vector_type(8)));
typedef _Float16 h16 __attribute__((ext_vector_type(16)));
typedef float    f8  __attribute__((ext_vector_type(8)));
typedef int      v4i __attribute__((vector_size(16)));

// Problem constants (from reference)
#define BATCH 4
#define TQ    2048
#define TKV   2048
#define QDIM  1024
#define KVDIM 768
#define HEADS 16
#define DK    64

// ---- CDNA5 async global->LDS copy (ASYNCcnt-tracked), with fallback ----
#if defined(__has_builtin)
#if __has_builtin(__builtin_amdgcn_global_load_async_to_lds_b128) && \
    __has_builtin(__builtin_amdgcn_s_wait_asynccnt)
#define CA_ASYNC 1
#else
#define CA_ASYNC 0
#endif
#else
#define CA_ASYNC 0
#endif

// clang prints addrspace(1) as __device__ in HIP: builtin takes v4i AS1* src.
typedef __attribute__((address_space(1))) v4i as1_v4i;
typedef __attribute__((address_space(3))) v4i as3_v4i;

// copy 16 bytes (8 halves) global -> LDS
static __device__ inline void lds_copy16(const _Float16* g, _Float16* l) {
#if CA_ASYNC
    __builtin_amdgcn_global_load_async_to_lds_b128((as1_v4i*)g, (as3_v4i*)l, 0, 0);
#else
    *(h8*)l = *(const h8*)g;
#endif
}
static __device__ inline void async_wait0() {
#if CA_ASYNC
    __builtin_amdgcn_s_wait_asynccnt(0);
#endif
}

static __device__ inline f8 wmma_f16(h16 a, h16 b, f8 c) {
    // D = A(16x32 f16) x B(32x16 f16) + C(16x16 f32)
    return __builtin_amdgcn_wmma_f32_16x16x32_f16(
        /*neg_a=*/false, a, /*neg_b=*/false, b,
        /*c_mod=*/(short)0, c, /*reuse_a=*/false, /*reuse_b=*/false);
}

// Load a 16x32 f16 A-fragment for this lane from a row-major row pointer.
// Contiguous k order used here: lane L holds row M=L&15; half=L>>4 selects K {0-15} vs {16-31}.
static __device__ inline h16 load_a_frag(const _Float16* __restrict__ rowk0, int half) {
    h8 lo = *(const h8*)(rowk0 + half * 16);
    h8 hi = *(const h8*)(rowk0 + half * 16 + 8);
    h16 a;
#pragma unroll
    for (int i = 0; i < 8; ++i) { a[i] = lo[i]; a[8 + i] = hi[i]; }
    return a;
}

// ---------------- fp32 -> fp16 convert ----------------
__global__ void ca_cvt_f32_f16(const float* __restrict__ src,
                               _Float16* __restrict__ dst, int n) {
    int i = blockIdx.x * blockDim.x + threadIdx.x;
    int stride = gridDim.x * blockDim.x;
    for (; i < n; i += stride) dst[i] = (_Float16)src[i];
}

// ---------------- GEMM: Y[M][N] = X[M][K] @ W[N][K]^T ----------------
// 256 threads = 8 waves; block tile 128(M) x 64(N); wave tile 16 x 64.
// B tile (64 cols x 32 k = 4KB) staged in LDS via async copy, double-buffered.
// mode 0: fp16 row-major; mode 1: fp16 scattered as Vt[(b*H+h)*DK+d][tok];
// mode 2: fp32 row-major.
__global__ __launch_bounds__(256)
void ca_gemm_xwT(const _Float16* __restrict__ X, const _Float16* __restrict__ W,
                 _Float16* __restrict__ Yh, float* __restrict__ Yf,
                 int M, int N, int K, int mode) {
    const int tid  = threadIdx.x;
    const int lane = tid & 31;
    const int wave = tid >> 5;
    const int half = lane >> 4;
    const int nl   = lane & 15;

    const int rowBase = blockIdx.y * 128 + wave * 16;
    const int colBase = blockIdx.x * 64;

    __shared__ __align__(16) _Float16 sB[2][64][32];

    const _Float16* xrow = X + (size_t)(rowBase + nl) * K;

    // cooperative B-tile copy: 256 threads x 16B = 4KB = 64 rows x 32 halves
    const int brow   = tid >> 2;
    const int bchunk = (tid & 3) * 8;
    const _Float16* wbase = W + (size_t)(colBase + brow) * K + bchunk;

    lds_copy16(wbase, &sB[0][brow][bchunk]);  // k0 = 0 into buffer 0

    f8 c[4] = {};
    int buf = 0;
    for (int k0 = 0; k0 < K; k0 += 32) {
        async_wait0();
        __syncthreads();
        if (k0 + 32 < K) {
            lds_copy16(wbase + k0 + 32, &sB[buf ^ 1][brow][bchunk]);
            __builtin_prefetch(xrow + k0 + 32, 0, 3);
        }
        h16 a = load_a_frag(xrow + k0, half);
#pragma unroll
        for (int t = 0; t < 4; ++t) {
            h16 b = *(const h16*)&sB[buf][t * 16 + nl][half * 16];
            c[t] = wmma_f16(a, b, c[t]);
        }
        buf ^= 1;
    }

#pragma unroll
    for (int t = 0; t < 4; ++t) {
#pragma unroll
        for (int r = 0; r < 8; ++r) {
            int row = rowBase + r + 8 * half;   // C/D layout: vgpr r, lane half
            int col = colBase + t * 16 + nl;
            float v = c[t][r];
            if (mode == 0) {
                Yh[(size_t)row * N + col] = (_Float16)v;
            } else if (mode == 1) {
                // scatter V as Vt[(b*H+h)*DK + d][tok] for contiguous PV B-loads
                int b   = row >> 11;          // / TKV
                int tok = row & (TKV - 1);
                int h   = col >> 6;           // / DK
                int d   = col & (DK - 1);
                Yh[(((size_t)((b << 4) + h) * DK + d) << 11) + tok] = (_Float16)v;
            } else {
                Yf[(size_t)row * N + col] = v;
            }
        }
    }
}

// ---------------- Flash attention ----------------
// grid: (TQ/64, BATCH*HEADS); block: 128 (4 waves). Wave owns 16 q rows.
// K and V^T tiles shared by all waves -> staged in LDS via async copy,
// double-buffered. Qp,Kp: [B*T][QDIM] fp16. Vt: [(b*H+h)*DK+d][TKV] fp16.
__global__ __launch_bounds__(128)
void ca_flash_attn(const _Float16* __restrict__ Qp,
                   const _Float16* __restrict__ Kp,
                   const _Float16* __restrict__ Vt,
                   _Float16* __restrict__ AO) {
    const int tid  = threadIdx.x;
    const int lane = tid & 31;
    const int wave = tid >> 5;
    const int half = lane >> 4;
    const int nl   = lane & 15;

    const int bh = blockIdx.y;
    const int b  = bh >> 4;
    const int h  = bh & 15;
    const int qBase = blockIdx.x * 64 + wave * 16;
    const float scale = 0.125f;   // 1/sqrt(64)

    __shared__ __align__(16) _Float16 sK[2][32][64];  // [kv][d]
    __shared__ __align__(16) _Float16 sV[2][64][32];  // [d][kv]
    __shared__ __align__(16) _Float16 sP[4][16][32];

    // cooperative K-tile copy: 32 rows x 128B; 4 threads/row, 32B each
    const int kr = tid >> 2, kc = (tid & 3) * 16;
    const _Float16* ksrc0 = Kp + (size_t)(b * TKV + kr) * QDIM + h * DK + kc;
    // cooperative V-tile copy: 64 rows x 64B; 2 threads/row, 32B each
    const int vr = tid >> 1, vc = (tid & 1) * 16;
    const _Float16* vsrc0 = Vt + (size_t)(bh * DK + vr) * TKV + vc;

    // Q A-fragments (d = 0..31 and 32..63), loaded once
    const _Float16* qrow = Qp + (size_t)(b * TQ + qBase + nl) * QDIM + h * DK;
    h16 aq0 = load_a_frag(qrow, half);
    h16 aq1 = load_a_frag(qrow + 32, half);

    // issue kv0 = 0 into buffer 0
    {
        const _Float16* ks = ksrc0;  // kv offset 0
        lds_copy16(ks,     &sK[0][kr][kc]);
        lds_copy16(ks + 8, &sK[0][kr][kc + 8]);
        const _Float16* vs = vsrc0;
        lds_copy16(vs,     &sV[0][vr][vc]);
        lds_copy16(vs + 8, &sV[0][vr][vc + 8]);
    }

    float m_r[8], l_r[8];
#pragma unroll
    for (int r = 0; r < 8; ++r) { m_r[r] = -3.0e38f; l_r[r] = 0.0f; }
    f8 acc[4] = {};

    int buf = 0;
    for (int kv0 = 0; kv0 < TKV; kv0 += 32) {
        async_wait0();
        __syncthreads();
        if (kv0 + 32 < TKV) {
            const _Float16* ks = ksrc0 + (size_t)(kv0 + 32) * QDIM;
            lds_copy16(ks,     &sK[buf ^ 1][kr][kc]);
            lds_copy16(ks + 8, &sK[buf ^ 1][kr][kc + 8]);
            const _Float16* vs = vsrc0 + kv0 + 32;
            lds_copy16(vs,     &sV[buf ^ 1][vr][vc]);
            lds_copy16(vs + 8, &sV[buf ^ 1][vr][vc + 8]);
        }

        // ---- scores: S[16 q][32 kv] from LDS K tile ----
        f8 s[2] = {};
#pragma unroll
        for (int u = 0; u < 2; ++u) {
            h16 bk0 = *(const h16*)&sK[buf][u * 16 + nl][half * 16];
            h16 bk1 = *(const h16*)&sK[buf][u * 16 + nl][32 + half * 16];
            s[u] = wmma_f16(aq0, bk0, s[u]);
            s[u] = wmma_f16(aq1, bk1, s[u]);
        }

        // ---- online softmax (rows live in 16-lane half-wave groups) ----
#pragma unroll
        for (int r = 0; r < 8; ++r) {
            float x0 = s[0][r] * scale;
            float x1 = s[1][r] * scale;
            float mx = fmaxf(x0, x1);
#pragma unroll
            for (int off = 8; off >= 1; off >>= 1)
                mx = fmaxf(mx, __shfl_xor(mx, off, 16));
            float mnew = fmaxf(m_r[r], mx);
            float cf = __expf(m_r[r] - mnew);
            float p0 = __expf(x0 - mnew);
            float p1 = __expf(x1 - mnew);
            float ps = p0 + p1;
#pragma unroll
            for (int off = 8; off >= 1; off >>= 1)
                ps += __shfl_xor(ps, off, 16);
            l_r[r] = l_r[r] * cf + ps;
            m_r[r] = mnew;
#pragma unroll
            for (int t = 0; t < 4; ++t) acc[t][r] *= cf;
            // C-layout -> LDS (row r+8*half, cols nl and 16+nl)
            sP[wave][r + 8 * half][nl]      = (_Float16)p0;
            sP[wave][r + 8 * half][16 + nl] = (_Float16)p1;
        }

        // ---- re-swizzle P into A-fragment layout (same-wave LDS, in order) ----
        h16 ap;
#pragma unroll
        for (int i = 0; i < 8; ++i) {
            ap[i]     = sP[wave][nl][half * 16 + i];
            ap[8 + i] = sP[wave][nl][half * 16 + 8 + i];
        }

        // ---- O += P(16x32) x V(32x64) from LDS V tile ----
#pragma unroll
        for (int t = 0; t < 4; ++t) {
            h16 bv = *(const h16*)&sV[buf][t * 16 + nl][half * 16];
            acc[t] = wmma_f16(ap, bv, acc[t]);
        }
        buf ^= 1;
    }

    // ---- epilogue: normalize, store token-major fp16 ----
#pragma unroll
    for (int t = 0; t < 4; ++t) {
#pragma unroll
        for (int r = 0; r < 8; ++r) {
            int q = qBase + r + 8 * half;
            float v = acc[t][r] / l_r[r];
            AO[(size_t)(b * TQ + q) * QDIM + h * DK + t * 16 + nl] = (_Float16)v;
        }
    }
}

// ---------------- host launch ----------------
extern "C" void kernel_launch(void* const* d_in, const int* in_sizes, int n_in,
                              void* d_out, int out_size, void* d_ws, size_t ws_size,
                              hipStream_t stream) {
    const float* q_tok = (const float*)d_in[0];   // [4,2048,1024]
    const float* kv_tok = (const float*)d_in[1];  // [4,2048,768]
    const float* Wq = (const float*)d_in[2];      // [1024,1024]
    const float* Wk = (const float*)d_in[3];      // [1024,768]
    const float* Wv = (const float*)d_in[4];      // [1024,768]
    const float* Wo = (const float*)d_in[5];      // [1024,1024]
    float* out = (float*)d_out;

    _Float16* ws = (_Float16*)d_ws;
    const size_t N_Q16  = (size_t)BATCH * TQ * QDIM;    // 8388608
    const size_t N_KV16 = (size_t)BATCH * TKV * KVDIM;  // 6291456
    const size_t N_WQ = (size_t)QDIM * QDIM;            // 1048576
    const size_t N_WK = (size_t)QDIM * KVDIM;           // 786432

    size_t off = 0;
    _Float16* q16  = ws + off; off += N_Q16;
    _Float16* kv16 = ws + off; off += N_KV16;
    _Float16* wq16 = ws + off; off += N_WQ;
    _Float16* wk16 = ws + off; off += N_WK;
    _Float16* wv16 = ws + off; off += N_WK;
    _Float16* wo16 = ws + off; off += N_WQ;
    _Float16* Qp   = ws + off; off += N_Q16;
    _Float16* Kp   = ws + off; off += N_Q16;
    _Float16* Vt   = ws + off; off += N_Q16;
    _Float16* AO   = ws + off; off += N_Q16;

    // 1) fp32 -> fp16 conversions
    auto cvt = [&](const float* s, _Float16* d, int n) {
        int blocks = (n + 1023) / 1024;
        ca_cvt_f32_f16<<<blocks, 256, 0, stream>>>(s, d, n);
    };
    cvt(q_tok,  q16,  (int)N_Q16);
    cvt(kv_tok, kv16, (int)N_KV16);
    cvt(Wq, wq16, (int)N_WQ);
    cvt(Wk, wk16, (int)N_WK);
    cvt(Wv, wv16, (int)N_WK);
    cvt(Wo, wo16, (int)N_WQ);

    const int M = BATCH * TQ;  // 8192
    dim3 gblk(256);
    dim3 ggrid(QDIM / 64, M / 128);  // (16, 64)

    // 2) projections
    ca_gemm_xwT<<<ggrid, gblk, 0, stream>>>(q16,  wq16, Qp, nullptr, M, QDIM, QDIM,  0);
    ca_gemm_xwT<<<ggrid, gblk, 0, stream>>>(kv16, wk16, Kp, nullptr, M, QDIM, KVDIM, 0);
    ca_gemm_xwT<<<ggrid, gblk, 0, stream>>>(kv16, wv16, Vt, nullptr, M, QDIM, KVDIM, 1);

    // 3) flash attention
    dim3 fgrid(TQ / 64, BATCH * HEADS);  // (32, 64)
    ca_flash_attn<<<fgrid, dim3(128), 0, stream>>>(Qp, Kp, Vt, AO);

    // 4) output projection (fp32 out)
    ca_gemm_xwT<<<ggrid, gblk, 0, stream>>>(AO, wo16, nullptr, out, M, QDIM, QDIM, 2);
}
